// NCSSD_10737418240376
// MI455X (gfx1250) — compile-verified, matrix operand order checked by reading, not run on hardware
//
#include <hip/hip_runtime.h>
#include <hip/hip_bf16.h>

typedef __attribute__((ext_vector_type(16))) _Float16 v16h;
typedef __attribute__((ext_vector_type(8)))  _Float16 v8h;
typedef __attribute__((ext_vector_type(8)))  float    v8f;
typedef __attribute__((ext_vector_type(4)))  unsigned int u32x4;
typedef __attribute__((ext_vector_type(8)))  int          i32x8;
typedef __attribute__((ext_vector_type(4)))  int          i32x4;

#define LDS_STRIDE 40   // 32 K-halves + 8 pad (= TDM pad: 16 DW interval, 4 DW pad)

// ---------------------------------------------------------------------------
// TDM 2D tile load: global (f16, row-major, stride_elems) -> LDS with padding
// so each tile row lands at LDS_STRIDE halves. D# packed per ISA 8.3/8.4.
// Issue from ONE wave only (TDM ignores EXEC); caller waits TENSORcnt.
// ---------------------------------------------------------------------------
__device__ __forceinline__ void tdm_load_2d(unsigned lds_off, const void* gsrc,
                                            unsigned tile_d0, unsigned tile_d1,
                                            unsigned long stride_elems) {
  unsigned long ga = (unsigned long)gsrc;
  u32x4 g0;
  g0[0] = 1u;                                   // count=1, user descriptor
  g0[1] = lds_off;                              // lds_addr (bytes)
  g0[2] = (unsigned)(ga & 0xffffffffu);         // global_addr[31:0]
  g0[3] = (unsigned)((ga >> 32) & 0x01ffffffu)  // global_addr[56:32]
          | (2u << 30);                         // type=2 (image)
  i32x8 g1;
  g1[0] = (int)((1u << 16)      // data_size = 2 bytes
              | (1u << 20)      // pad_enable
              | (3u << 22)      // pad_interval: 16 DWORDs (= 64B tile row)
              | (3u << 25));    // pad_amount: 4 DWORDs (= 8 halves)
  g1[1] = (int)((tile_d0 & 0xffffu) << 16);                       // tensor_dim0 lo16
  g1[2] = (int)(((tile_d0 >> 16) & 0xffffu) | ((tile_d1 & 0xffffu) << 16)); // d0 hi / d1 lo
  g1[3] = (int)(((tile_d1 >> 16) & 0xffffu) | ((tile_d0 & 0xffffu) << 16)); // d1 hi / tile_dim0
  g1[4] = (int)(tile_d1 & 0xffffu);                               // tile_dim1 (tile_dim2=0)
  g1[5] = (int)(stride_elems & 0xffffffffu);                      // dim0_stride lo32
  g1[6] = (int)((stride_elems >> 32) & 0xffffu);                  // dim0_stride hi16
  g1[7] = 0;
  i32x4 z4; z4[0] = 0; z4[1] = 0; z4[2] = 0; z4[3] = 0;           // groups 2/3 unused (2D)
  i32x8 z8;
#pragma unroll
  for (int i = 0; i < 8; ++i) z8[i] = 0;
  __builtin_amdgcn_tensor_load_to_lds(g0, g1, z4, z4, z8, 0);
}

__device__ __forceinline__ unsigned lds_off_of(const void* p) {
  return (unsigned)(size_t)p;   // flat LDS pointer: low 32 bits = LDS offset
}

// 16x32 f16 fragment from LDS (row-major, LDS_STRIDE halves per row).
__device__ __forceinline__ v16h frag_ld(const _Float16* p) {
  const int lane = threadIdx.x & 31;
  const _Float16* q = p + (lane & 15) * LDS_STRIDE + ((lane >> 4) << 3);
  v8h lo = *(const v8h*)(q);
  v8h hi = *(const v8h*)(q + 16);
  v16h f;
#pragma unroll
  for (int e = 0; e < 8; ++e) { f[e] = lo[e]; f[e + 8] = hi[e]; }
  return f;
}

__device__ __forceinline__ v8f wmma32(const v16h& a, const v16h& b, const v8f& c) {
  return __builtin_amdgcn_wmma_f32_16x16x32_f16(false, a, false, b, (short)0, c,
                                                false, false);
}

__device__ __forceinline__ float silu(float v) { return v / (1.0f + __expf(-v)); }

// ===========================================================================
// Packing kernels (fp32 -> f16, guard-free GEMM tiles)
// ===========================================================================
__global__ void pack_inproj_w(const float* __restrict__ w, _Float16* __restrict__ wh) {
  int i = blockIdx.x * 256 + threadIdx.x;      // 1280*256, rows >=1160 zero-padded
  if (i >= 1280 * 256) return;
  int r = i >> 8;
  wh[i] = (r < 1160) ? (_Float16)w[i] : (_Float16)0.f;
}

__global__ void pack_f32_to_f16(const float* __restrict__ x, _Float16* __restrict__ y,
                                int n) {
  int i = blockIdx.x * 256 + threadIdx.x;
  if (i < n) y[i] = (_Float16)x[i];
}

// out_proj_w (oc,ic,kh,kw) fp32 -> tap-major f16 [9][256][512]
__global__ void reorder_convw(const float* __restrict__ w, _Float16* __restrict__ wh) {
  int i = blockIdx.x * 256 + threadIdx.x;
  if (i >= 9 * 256 * 512) return;
  int tap = i / (256 * 512), r = i % (256 * 512);
  int oc = r / 512, ic = r % 512;
  wh[i] = (_Float16)w[((size_t)(oc * 512 + ic) * 3 + tap / 3) * 3 + tap % 3];
}

// ===========================================================================
// 1) in_proj GEMM: Z[b,m,n] = sum_k Wh[m,k]*Uh[b,k,n].
//    M-tile 256 (2 strips/wave, 8 WMMA/step), ping-pong LDS, TDM A-tiles.
// ===========================================================================
__global__ __launch_bounds__(256) void gemm_inproj(
    const _Float16* __restrict__ Wh, const _Float16* __restrict__ Uh,
    float* __restrict__ Z, int M, int K, int L) {
  __shared__ __align__(16) _Float16 As[2][256 * LDS_STRIDE];
  __shared__ __align__(16) _Float16 Bs[2][64 * LDS_STRIDE];
  const int tid = threadIdx.x, wave = tid >> 5, lane = tid & 31;
  const int b  = blockIdx.z;
  const int m0 = blockIdx.y * 256;
  const int n0 = blockIdx.x * 64;
  const _Float16* Ub = Uh + (size_t)b * K * L;
  const int kk = tid >> 3, nseg = tid & 7;   // Bt staging coords (32k x 8 segs)

  v8f acc[2][4];
#pragma unroll
  for (int s = 0; s < 2; ++s)
#pragma unroll
    for (int i = 0; i < 4; ++i)
#pragma unroll
      for (int e = 0; e < 8; ++e) acc[s][i][e] = 0.0f;

  // prologue: tile k0=0 -> buffer 0
  if (wave == 0)
    tdm_load_2d(lds_off_of(As[0]), Wh + (size_t)m0 * K, 32, 256, (unsigned long)K);
  {
    v8h v = *(const v8h*)(Ub + (size_t)kk * L + n0 + nseg * 8);
#pragma unroll
    for (int e = 0; e < 8; ++e) Bs[0][(nseg * 8 + e) * LDS_STRIDE + kk] = v[e];
  }
  if (wave == 0) __builtin_amdgcn_s_wait_tensorcnt(0);
  __syncthreads();

  for (int k0 = 0; k0 < K; k0 += 32) {
    const int cu = (k0 >> 5) & 1, nx = cu ^ 1;
    const bool more = (k0 + 32) < K;
    v8h breg;
#pragma unroll
    for (int e = 0; e < 8; ++e) breg[e] = (_Float16)0.f;
    if (more) {     // issue next tile's TDM + global loads BEFORE the math
      if (wave == 0)
        tdm_load_2d(lds_off_of(As[nx]), Wh + (size_t)m0 * K + k0 + 32, 32, 256,
                    (unsigned long)K);
      breg = *(const v8h*)(Ub + (size_t)(k0 + 32 + kk) * L + n0 + nseg * 8);
    }
    v16h a0 = frag_ld(As[cu] + wave * 16 * LDS_STRIDE);
    v16h a1 = frag_ld(As[cu] + (wave + 8) * 16 * LDS_STRIDE);
#pragma unroll
    for (int nb = 0; nb < 4; ++nb) {
      v16h bf = frag_ld(Bs[cu] + nb * 16 * LDS_STRIDE);
      acc[0][nb] = wmma32(a0, bf, acc[0][nb]);
      acc[1][nb] = wmma32(a1, bf, acc[1][nb]);
    }
    if (more) {     // store staged regs after math (loadcnt wait overlaps WMMA)
#pragma unroll
      for (int e = 0; e < 8; ++e) Bs[nx][(nseg * 8 + e) * LDS_STRIDE + kk] = breg[e];
    }
    if (wave == 0) __builtin_amdgcn_s_wait_tensorcnt(0);
    __syncthreads();
  }
#pragma unroll
  for (int sp = 0; sp < 2; ++sp) {
    const int rbase = m0 + (wave + sp * 8) * 16 + ((lane >> 4) << 3);
    const int cbase = n0 + (lane & 15);
#pragma unroll
    for (int nb = 0; nb < 4; ++nb)
#pragma unroll
      for (int i = 0; i < 8; ++i) {
        int gm = rbase + i;
        if (gm < M) Z[((size_t)b * M + gm) * L + cbase + nb * 16] = acc[sp][nb][i];
      }
  }
}

// ===========================================================================
// 2) Depthwise causal conv1d + SiLU (OutT = f16 for xBC, f32 for z-gate)
// ===========================================================================
template <typename OutT>
__global__ void dwconv_silu(const float* __restrict__ x, const float* __restrict__ w,
                            const float* __restrict__ bias, OutT* __restrict__ y,
                            int K, int L, long xbstride, long ybstride) {
  const int b = blockIdx.y, c = blockIdx.x;
  const float* xr = x + (size_t)b * xbstride + (size_t)c * L;
  OutT* yr = y + (size_t)b * ybstride + (size_t)c * L;
  float wl[4];
  for (int k = 0; k < K; ++k) wl[k] = w[c * K + k];
  const float bb = bias[c];
  for (int l = threadIdx.x; l < L; l += blockDim.x) {
    float s = bb;
    for (int k = 0; k < K; ++k) {
      int j = l - (K - 1) + k;
      if (j >= 0) s += wl[k] * xr[j];
    }
    yr[l] = (OutT)silu(s);
  }
}

// 2b) dt conv (K=4) -> dA = silu(conv)*(-exp(A_log))
__global__ void dt_kernel(const float* __restrict__ x, const float* __restrict__ w,
                          const float* __restrict__ bias, const float* __restrict__ A_log,
                          float* __restrict__ dAout, int L, long xbstride) {
  const int b = blockIdx.y, h = blockIdx.x;
  const float* xr = x + (size_t)b * xbstride + (size_t)h * L;
  const float Aneg = -__expf(A_log[h]);
  float wl[4];
  for (int k = 0; k < 4; ++k) wl[k] = w[h * 4 + k];
  const float bb = bias[h];
  for (int l = threadIdx.x; l < L; l += blockDim.x) {
    float s = bb;
    for (int k = 0; k < 4; ++k) {
      int j = l - 3 + k;
      if (j >= 0) s += wl[k] * xr[j];
    }
    dAout[((size_t)b * 8 + h) * L + l] = silu(s) * Aneg;
  }
}

// 2c) Vs[b,c,l] = x[b,c,l] * dA[b,h(c),l]  (f16, makes KV B-operand TDM-able)
__global__ void vscale_kernel(const _Float16* __restrict__ xch,
                              const float* __restrict__ dA,
                              _Float16* __restrict__ Vsh, int L) {
  const int b = blockIdx.y, c = blockIdx.x, h = c >> 6;
  const _Float16* xr = xch + ((size_t)b * 640 + c) * L;
  const float* ar = dA + ((size_t)b * 8 + h) * L;
  _Float16* vr = Vsh + ((size_t)b * 512 + c) * L;
  for (int l = threadIdx.x; l < L; l += blockDim.x)
    vr[l] = (_Float16)((float)xr[l] * ar[l]);
}

// ===========================================================================
// 3) KV[b,h,n,d] = sum_l Bm[n,l]*Vs[d,l] — both operands TDM, ping-pong LDS.
// ===========================================================================
__global__ __launch_bounds__(256) void kv_kernel(
    const _Float16* __restrict__ xch, const _Float16* __restrict__ Vsh,
    float* __restrict__ KV, int L) {
  __shared__ __align__(16) _Float16 As[2][64 * LDS_STRIDE];
  __shared__ __align__(16) _Float16 Bs[2][64 * LDS_STRIDE];
  const int bh = blockIdx.x, b = bh >> 3, h = bh & 7;
  const int tid = threadIdx.x, wave = tid >> 5, lane = tid & 31;
  const _Float16* Bm = xch + ((size_t)b * 640 + 512) * L;
  const _Float16* Vh = Vsh + ((size_t)b * 512 + h * 64) * L;

  v8f acc[2];
#pragma unroll
  for (int p = 0; p < 2; ++p)
#pragma unroll
    for (int e = 0; e < 8; ++e) acc[p][e] = 0.0f;

  if (wave == 0) {
    tdm_load_2d(lds_off_of(As[0]), Bm, 32, 64, (unsigned long)L);
    tdm_load_2d(lds_off_of(Bs[0]), Vh, 32, 64, (unsigned long)L);
    __builtin_amdgcn_s_wait_tensorcnt(0);
  }
  __syncthreads();

  for (int l0 = 0; l0 < L; l0 += 32) {
    const int cu = (l0 >> 5) & 1, nx = cu ^ 1;
    if ((l0 + 32) < L && wave == 0) {
      tdm_load_2d(lds_off_of(As[nx]), Bm + l0 + 32, 32, 64, (unsigned long)L);
      tdm_load_2d(lds_off_of(Bs[nx]), Vh + l0 + 32, 32, 64, (unsigned long)L);
    }
#pragma unroll
    for (int p = 0; p < 2; ++p) {
      int c = wave + p * 8;
      v16h a  = frag_ld(As[cu] + (c >> 2) * 16 * LDS_STRIDE);
      v16h bf = frag_ld(Bs[cu] + (c & 3) * 16 * LDS_STRIDE);
      acc[p] = wmma32(a, bf, acc[p]);
    }
    if (wave == 0) __builtin_amdgcn_s_wait_tensorcnt(0);
    __syncthreads();
  }
#pragma unroll
  for (int p = 0; p < 2; ++p) {
    int c = wave + p * 8;
    int m = (c >> 2) * 16 + ((lane >> 4) << 3);
    int n = (c & 3) * 16 + (lane & 15);
#pragma unroll
    for (int i = 0; i < 8; ++i)
      KV[((size_t)bh * 64 + m + i) * 64 + n] = acc[p][i];
  }
}

// ===========================================================================
// 4) y = Cm x KV + D*V, gated with silu(zc). K=64: both K-steps staged into
//    separate buffers up front -> single barrier for the whole GEMM.
// ===========================================================================
__global__ __launch_bounds__(256) void ygate_kernel(
    const _Float16* __restrict__ xch, const float* __restrict__ KV,
    const float* __restrict__ zsilu, const float* __restrict__ Dp,
    float* __restrict__ yg, int L) {
  __shared__ __align__(16) _Float16 As[2][128 * LDS_STRIDE];
  __shared__ __align__(16) _Float16 Bs[2][64 * LDS_STRIDE];
  const int bh = blockIdx.y, b = bh >> 3, h = bh & 7;
  const int tid = threadIdx.x, wave = tid >> 5, lane = tid & 31;
  const int l0 = blockIdx.x * 128;
  const _Float16* Cm = xch + ((size_t)b * 640 + 576) * L;  // 64 n-rows, l-contig
  const float* KVb = KV + (size_t)bh * 64 * 64;

  v8f acc[4];
#pragma unroll
  for (int i = 0; i < 4; ++i)
#pragma unroll
    for (int e = 0; e < 8; ++e) acc[i][e] = 0.0f;

#pragma unroll
  for (int st = 0; st < 2; ++st) {
    const int k0 = st * 32;
#pragma unroll
    for (int it = 0; it < 2; ++it) {           // A[m=l][k=n] transpose, v8h loads
      int idx = tid + it * 256;
      int kk = idx >> 4, mseg = idx & 15;
      v8h v = *(const v8h*)(Cm + (size_t)(k0 + kk) * L + l0 + mseg * 8);
#pragma unroll
      for (int e = 0; e < 8; ++e) As[st][(mseg * 8 + e) * LDS_STRIDE + kk] = v[e];
    }
    for (int idx = tid; idx < 512; idx += 256) {  // Bt[d][n] = KV[n][d]
      int kk = idx >> 4, dseg = idx & 15;
      float4 v = *(const float4*)(KVb + (size_t)(k0 + kk) * 64 + dseg * 4);
      Bs[st][(dseg * 4 + 0) * LDS_STRIDE + kk] = (_Float16)v.x;
      Bs[st][(dseg * 4 + 1) * LDS_STRIDE + kk] = (_Float16)v.y;
      Bs[st][(dseg * 4 + 2) * LDS_STRIDE + kk] = (_Float16)v.z;
      Bs[st][(dseg * 4 + 3) * LDS_STRIDE + kk] = (_Float16)v.w;
    }
  }
  __syncthreads();
#pragma unroll
  for (int st = 0; st < 2; ++st) {
    v16h a  = frag_ld(As[st] + wave * 16 * LDS_STRIDE);
    v16h b0 = frag_ld(Bs[st]);
    v16h b1 = frag_ld(Bs[st] + 16 * LDS_STRIDE);
    v16h b2 = frag_ld(Bs[st] + 32 * LDS_STRIDE);
    v16h b3 = frag_ld(Bs[st] + 48 * LDS_STRIDE);
    acc[0] = wmma32(a, b0, acc[0]);
    acc[1] = wmma32(a, b1, acc[1]);
    acc[2] = wmma32(a, b2, acc[2]);
    acc[3] = wmma32(a, b3, acc[3]);
  }
  const float Dh = Dp[h];
  const int lbase = l0 + wave * 16 + ((lane >> 4) << 3);
#pragma unroll
  for (int nb = 0; nb < 4; ++nb) {
    int c = h * 64 + nb * 16 + (lane & 15);
#pragma unroll
    for (int i = 0; i < 8; ++i) {
      int l = lbase + i;
      float V  = (float)xch[((size_t)b * 640 + c) * L + l];
      float zs = zsilu[((size_t)b * 512 + c) * L + l];
      yg[((size_t)b * L + l) * 512 + c] = (acc[nb][i] + Dh * V) * zs;
    }
  }
}

// ===========================================================================
// 5) RMSNorm over 512 channels per (b,l) -> f16 channel-major image.
// ===========================================================================
__global__ __launch_bounds__(256) void rmsnorm_kernel(
    const float* __restrict__ yg, const float* __restrict__ nw,
    _Float16* __restrict__ yimg, int L) {
  __shared__ float red[256];
  const int bl = blockIdx.x, b = bl >> 12, l = bl & 4095;
  const int tid = threadIdx.x;
  const float* row = yg + (size_t)bl * 512;
  float v0 = row[tid], v1 = row[tid + 256];
  red[tid] = v0 * v0 + v1 * v1;
  __syncthreads();
  for (int st = 128; st > 0; st >>= 1) {
    if (tid < st) red[tid] += red[tid + st];
    __syncthreads();
  }
  const float rstd = rsqrtf(red[0] * (1.0f / 512.0f) + 1e-5f);
  yimg[((size_t)b * 512 + tid) * L + l]       = (_Float16)(v0 * rstd * nw[tid]);
  yimg[((size_t)b * 512 + tid + 256) * L + l] = (_Float16)(v1 * rstd * nw[tid + 256]);
}

// ===========================================================================
// 6) 3x3 SAME conv: 144 K-steps (9 taps x 16), M=256 in one block
//    (2 strips/wave), ping-pong LDS, TDM A-tiles, register-staged B gathers.
// ===========================================================================
__global__ __launch_bounds__(256) void conv_gemm(
    const _Float16* __restrict__ Wk, const _Float16* __restrict__ yimg,
    float* __restrict__ out, int L) {
  __shared__ __align__(16) _Float16 As[2][256 * LDS_STRIDE];
  __shared__ __align__(16) _Float16 Bs[2][64 * LDS_STRIDE];
  const int tid = threadIdx.x, wave = tid >> 5, lane = tid & 31;
  const int b  = blockIdx.y;
  const int n0 = blockIdx.x * 64;
  const _Float16* Y = yimg + (size_t)b * 512 * L;

  v8f acc[2][4];
#pragma unroll
  for (int s = 0; s < 2; ++s)
#pragma unroll
    for (int i = 0; i < 4; ++i)
#pragma unroll
      for (int e = 0; e < 8; ++e) acc[s][i][e] = 0.0f;

  auto gatherB = [&](int tap, int k0, _Float16* breg) {
    const int dh = tap / 3 - 1, dw = tap % 3 - 1;
#pragma unroll
    for (int j = 0; j < 8; ++j) {
      int idx = tid + j * 256;
      int nn = idx >> 5, kq = idx & 31;
      int n = n0 + nn, hh = n >> 6, ww = n & 63;
      int sh = hh + dh, sw = ww + dw;
      breg[j] = ((unsigned)sh < 64u && (unsigned)sw < 64u)
                    ? Y[(size_t)(k0 + kq) * L + sh * 64 + sw]
                    : (_Float16)0.f;
    }
  };
  auto storeB = [&](int buf, const _Float16* breg) {
#pragma unroll
    for (int j = 0; j < 8; ++j) {
      int idx = tid + j * 256;
      int nn = idx >> 5, kq = idx & 31;
      Bs[buf][nn * LDS_STRIDE + kq] = breg[j];
    }
  };

  // prologue: step 0 (tap 0, k0 0) -> buffer 0
  if (wave == 0) tdm_load_2d(lds_off_of(As[0]), Wk, 32, 256, 512ul);
  {
    _Float16 breg[8];
    gatherB(0, 0, breg);
    storeB(0, breg);
  }
  if (wave == 0) __builtin_amdgcn_s_wait_tensorcnt(0);
  __syncthreads();

  for (int s = 0; s < 144; ++s) {
    const int cu = s & 1, nx = cu ^ 1;
    const bool more = (s + 1) < 144;
    _Float16 breg[8];
#pragma unroll
    for (int j = 0; j < 8; ++j) breg[j] = (_Float16)0.f;
    if (more) {
      const int tap_n = (s + 1) >> 4, k0_n = ((s + 1) & 15) << 5;
      if (wave == 0)
        tdm_load_2d(lds_off_of(As[nx]), Wk + (size_t)tap_n * 256 * 512 + k0_n,
                    32, 256, 512ul);
      gatherB(tap_n, k0_n, breg);
    }
    v16h a0 = frag_ld(As[cu] + wave * 16 * LDS_STRIDE);
    v16h a1 = frag_ld(As[cu] + (wave + 8) * 16 * LDS_STRIDE);
#pragma unroll
    for (int nb = 0; nb < 4; ++nb) {
      v16h bf = frag_ld(Bs[cu] + nb * 16 * LDS_STRIDE);
      acc[0][nb] = wmma32(a0, bf, acc[0][nb]);
      acc[1][nb] = wmma32(a1, bf, acc[1][nb]);
    }
    if (more) storeB(nx, breg);
    if (wave == 0) __builtin_amdgcn_s_wait_tensorcnt(0);
    __syncthreads();
  }
#pragma unroll
  for (int sp = 0; sp < 2; ++sp) {
    const int m = (wave + sp * 8) * 16 + ((lane >> 4) << 3);
#pragma unroll
    for (int nb = 0; nb < 4; ++nb) {
      int n = n0 + nb * 16 + (lane & 15);
#pragma unroll
      for (int i = 0; i < 8; ++i)
        out[((size_t)b * 256 + m + i) * L + n] = acc[sp][nb][i];
    }
  }
}

// ===========================================================================
extern "C" void kernel_launch(void* const* d_in, const int* in_sizes, int n_in,
                              void* d_out, int out_size, void* d_ws, size_t ws_size,
                              hipStream_t stream) {
  (void)in_sizes; (void)n_in; (void)out_size; (void)ws_size;
  const float* u          = (const float*)d_in[0];
  const float* in_proj_w  = (const float*)d_in[1];
  const float* conv_dt_w  = (const float*)d_in[2];
  const float* dt_bias    = (const float*)d_in[3];
  const float* conv_xBC_w = (const float*)d_in[4];
  const float* conv_xBC_b = (const float*)d_in[5];
  const float* conv_z_w   = (const float*)d_in[6];
  const float* conv_z_b   = (const float*)d_in[7];
  const float* A_log      = (const float*)d_in[8];
  const float* D_param    = (const float*)d_in[9];
  const float* norm_w     = (const float*)d_in[10];
  const float* out_proj_w = (const float*)d_in[11];
  float* out = (float*)d_out;

  const int B = 8, L = 4096, DIP = 1160, DM = 256;
  char* ws = (char*)d_ws;
  size_t off = 0;
  auto take = [&](size_t bytes) {
    size_t o = off;
    off += (bytes + 255) & ~(size_t)255;
    return o;
  };
  float*    zxbcdt = (float*)   (ws + take((size_t)B * DIP * L * 4));
  _Float16* Whp    = (_Float16*)(ws + take((size_t)1280 * 256 * 2));
  _Float16* uh     = (_Float16*)(ws + take((size_t)B * 256 * L * 2));
  _Float16* xch    = (_Float16*)(ws + take((size_t)B * 640 * L * 2));
  _Float16* Vsh    = (_Float16*)(ws + take((size_t)B * 512 * L * 2));
  float*    zsilu  = (float*)   (ws + take((size_t)B * 512 * L * 4));
  float*    dAbuf  = (float*)   (ws + take((size_t)B * 8 * L * 4));
  float*    KV     = (float*)   (ws + take((size_t)B * 8 * 64 * 64 * 4));
  float*    yg     = (float*)   (ws + take((size_t)B * L * 512 * 4));
  _Float16* yimg   = (_Float16*)(ws + take((size_t)B * 512 * L * 2));
  _Float16* wh     = (_Float16*)(ws + take((size_t)9 * 256 * 512 * 2));

  // 0) operand packing (f16, padded)
  pack_inproj_w<<<(1280 * 256 + 255) / 256, 256, 0, stream>>>(in_proj_w, Whp);
  pack_f32_to_f16<<<(B * 256 * L + 255) / 256, 256, 0, stream>>>(u, uh, B * 256 * L);
  reorder_convw<<<(9 * 256 * 512 + 255) / 256, 256, 0, stream>>>(out_proj_w, wh);
  // 1) in_proj GEMM (WMMA, TDM A-tiles, ping-pong)
  gemm_inproj<<<dim3(L / 64, 5, B), 256, 0, stream>>>(Whp, uh, zxbcdt, DIP, DM, L);
  // 2) depthwise convs + dt + V scaling
  dwconv_silu<_Float16><<<dim3(640, B), 256, 0, stream>>>(zxbcdt + (size_t)512 * L,
      conv_xBC_w, conv_xBC_b, xch, 3, L, (long)DIP * L, (long)640 * L);
  dwconv_silu<float><<<dim3(512, B), 256, 0, stream>>>(zxbcdt, conv_z_w,
      conv_z_b, zsilu, 3, L, (long)DIP * L, (long)512 * L);
  dt_kernel<<<dim3(8, B), 256, 0, stream>>>(zxbcdt + (size_t)1152 * L, conv_dt_w,
      dt_bias, A_log, dAbuf, L, (long)DIP * L);
  vscale_kernel<<<dim3(512, B), 256, 0, stream>>>(xch, dAbuf, Vsh, L);
  // 3) SSD state GEMM (WMMA, double-buffered TDM)
  kv_kernel<<<B * 8, 256, 0, stream>>>(xch, Vsh, KV, L);
  // 4) output GEMM + skip + gate (WMMA)
  ygate_kernel<<<dim3(L / 128, B * 8), 256, 0, stream>>>(xch, KV, zsilu, D_param, yg, L);
  // 5) RMSNorm -> f16 image
  rmsnorm_kernel<<<B * L, 256, 0, stream>>>(yg, norm_w, yimg, L);
  // 6) 3x3 conv implicit GEMM (WMMA, TDM A-tiles, ping-pong)
  conv_gemm<<<dim3(L / 64, B), 256, 0, stream>>>(wh, yimg, out, L);
}
